// GODIVA_18442589569945
// MI455X (gfx1250) — matrix operand order repaired and checked by reading, not compiled, not run on hardware
//
#include <hip/hip_runtime.h>

// ---------------------------------------------------------------------------
// CDNA5 / gfx1250 transformer forward pass (compile-only target).
// All GEMMs + attention contractions run through v_wmma_f32_16x16x32_f16
// (f16 inputs, f32 accumulate). Wave32 everywhere. GEMM tiles are staged
// through LDS with global_load_async_to_lds_b128/_b64 (ASYNCcnt-tracked DMA).
// ---------------------------------------------------------------------------

typedef __attribute__((ext_vector_type(16))) _Float16 v16h;
typedef __attribute__((ext_vector_type(8)))  float    v8f;

#define D_MODEL 256
#define D_HEAD  32

__device__ __forceinline__ v8f wmma_f16(v16h a, v16h b, v8f c) {
  // (neg_a, A, neg_b, B, c_mod, C, reuse_a, reuse_b)
  return __builtin_amdgcn_wmma_f32_16x16x32_f16(false, a, false, b, (short)0, c,
                                                false, false);
}

// A-operand (16x32 f16) k order used here (contiguous, B matches): lanes 0-15 hold M=lane; halves i=0..15 are
// K = base+i, base = 16*(lane>=16).
__device__ __forceinline__ int a_kidx(int i, int hg) {
  return hg * 16 + i;
}

// ---- CDNA5 async global->LDS copies, ASYNCcnt tracked ---------------------
// VDST = per-lane LDS byte offset (low 32 bits of the generic shared address:
// LDS aperture places the LDS offset in addr[31:0]); VADDR = 64-bit global VA.
__device__ __forceinline__ void async_copy_b128(void* lds_ptr, const void* gptr) {
  unsigned lds_off = (unsigned)(size_t)lds_ptr;
  asm volatile("global_load_async_to_lds_b128 %0, %1, off"
               :: "v"(lds_off), "v"(gptr) : "memory");
}
__device__ __forceinline__ void async_copy_b64(void* lds_ptr, const void* gptr) {
  unsigned lds_off = (unsigned)(size_t)lds_ptr;
  asm volatile("global_load_async_to_lds_b64 %0, %1, off"
               :: "v"(lds_off), "v"(gptr) : "memory");
}
__device__ __forceinline__ void wait_async_zero() {
  asm volatile("s_wait_asynccnt 0" ::: "memory");
}

// ---------------------------------------------------------------------------
// Embedding * sqrt(D) + sinusoidal positional encoding (concat-sin/cos form).
// ---------------------------------------------------------------------------
__global__ void embed_pos_kernel(const int* __restrict__ toks,
                                 const float* __restrict__ emb,
                                 float* __restrict__ x, int S, int total) {
  int idx = blockIdx.x * blockDim.x + threadIdx.x;
  if (idx >= total) return;
  int d   = idx & (D_MODEL - 1);
  int row = idx >> 8;                 // flat (b*S + s)
  int s   = row % S;
  int tok = toks[row];
  float pe;
  if (d < D_MODEL / 2) {
    float ang = (float)s * __powf(10000.f, -(2.f * (float)d) / (float)D_MODEL);
    pe = __sinf(ang);
  } else {
    int dp = d - D_MODEL / 2;
    float ang = (float)s * __powf(10000.f, -(2.f * (float)dp) / (float)D_MODEL);
    pe = __cosf(ang);
  }
  x[idx] = emb[(size_t)tok * D_MODEL + d] * 16.0f /* sqrt(256) */ + pe;
}

// ---------------------------------------------------------------------------
// C[M,N] = act(A[M,K] @ W[K,N] + bias[N]).
// Block = 256 threads (8 waves) computing a 16x128 output tile; per k-step
// the block cooperatively stages A(16x32 f32, 2KB) and W(32x128 f32, 16KB)
// into LDS. A is always staged with async b128 DMA (K%4==0 at every call
// site). W path per block (uniform branch):
//   wpath 2: interior stripe, N%4==0  -> async b128 DMA
//   wpath 1: interior stripe, N%2==0  -> async b64 DMA (vocab GEMM, N=10002)
//   wpath 0: ragged tail              -> clamped scalar fill, one 64-bit row
//            base per thread (16 consecutive clamped cols per thread).
// Each wave then builds its WMMA fragments from LDS. K%32==0, M%16==0 always.
// ---------------------------------------------------------------------------
#define BN 128
__global__ __launch_bounds__(256)
void gemm_wmma_kernel(const float* __restrict__ A, const float* __restrict__ W,
                      const float* __restrict__ bias, float* __restrict__ C,
                      int M, int N, int K, int relu) {
  __shared__ float aLds[16 * 32];    //  2 KB
  __shared__ float wLds[32 * BN];    // 16 KB
  const int tid  = threadIdx.x;
  const int lane = tid & 31;
  const int wv   = tid >> 5;                 // wave id 0..7 -> N sub-tile
  const int m0   = blockIdx.y * 16;
  const int nblk = blockIdx.x * BN;
  const int hg   = lane >> 4;                // half-wave group
  const int l15  = lane & 15;
  const int wpath = (nblk + BN <= N) ? (((N & 3) == 0) ? 2
                                       : (((N & 1) == 0) ? 1 : 0))
                                     : 0;
  (void)M;

  v8f acc = {};
  for (int k0 = 0; k0 < K; k0 += 32) {
    __syncthreads();   // previous iteration's LDS reads complete
    // ---- A tile: async b128 DMA, 128 lanes x 16B ----
    if (tid < 128) {
      int r  = tid >> 3;                     // 0..15
      int c4 = (tid & 7) * 4;                // 0,4,..,28
      async_copy_b128(&aLds[r * 32 + c4],
                      A + (size_t)(m0 + r) * K + k0 + c4);
    }
    // ---- W tile ----
    if (wpath == 2) {
#pragma unroll
      for (int rr = 0; rr < 4; ++rr) {       // 256 thr x 4 x 16B
        int r  = (tid >> 5) + rr * 8;        // 0..31
        int c4 = (tid & 31) * 4;             // 0..124
        async_copy_b128(&wLds[r * BN + c4],
                        W + (size_t)(k0 + r) * N + nblk + c4);
      }
    } else if (wpath == 1) {
#pragma unroll
      for (int rr = 0; rr < 8; ++rr) {       // 256 thr x 8 x 8B
        int r  = (tid >> 6) + rr * 4;        // 0..31
        int c2 = (tid & 63) * 2;             // 0..126
        async_copy_b64(&wLds[r * BN + c2],
                       W + (size_t)(k0 + r) * N + nblk + c2);
      }
    } else {
      // ragged tail: one row segment per thread, single 64-bit row base
      int r    = tid >> 3;                   // 0..31
      int cseg = (tid & 7) * 16;             // 0..112
      const float* wr = W + (size_t)(k0 + r) * N;
      float* ld = &wLds[r * BN + cseg];
#pragma unroll
      for (int e = 0; e < 16; ++e) {
        int cg = nblk + cseg + e;
        cg = (cg < N) ? cg : (N - 1);        // clamp; store guarded later
        ld[e] = wr[cg];
      }
    }
    wait_async_zero();                       // drain this wave's ASYNCcnt
    __syncthreads();                         // all waves' tiles visible

    v16h av, bfrag;
#pragma unroll
    for (int i = 0; i < 16; ++i)
      av[i] = (_Float16)aLds[l15 * 32 + a_kidx(i, hg)];
    // B-matrix (32x16): lane = N-col within half, halves i -> K = 16*hg + i
#pragma unroll
    for (int i = 0; i < 16; ++i)
      bfrag[i] = (_Float16)wLds[(hg * 16 + i) * BN + wv * 16 + l15];
    acc = wmma_f16(av, bfrag, acc);
  }

  // C/D layout: vgpr j -> row m0 + j + 8*hg, col nblk + wv*16 + l15
  const int cn = nblk + wv * 16 + l15;
  if (cn < N) {
#pragma unroll
    for (int j = 0; j < 8; ++j) {
      int r = m0 + j + hg * 8;
      float v = acc[j] + bias[cn];
      if (relu) v = fmaxf(v, 0.f);
      C[(size_t)r * N + cn] = v;
    }
  }
}

// ---------------------------------------------------------------------------
// Streaming (flash-style) multi-head attention. One wave per (b, h, 16-query
// tile). Scores: QK^T with K-dim = DH = 32 (one WMMA per 16-key subtile).
// Online softmax with 16-lane shuffle row-reductions, P re-laid out through
// LDS into A-matrix format, then P@V via two more WMMAs (two 16-wide d tiles).
// mode 0: additive mask (1 - fmask[b,k]) * -1e9
// mode 1: reference's decoder mask: maxv(k>q, tok[k]==0), applied as
//         (1 - maxv) * -1e9 (quirk of the source preserved verbatim).
// Output written head-merged into [B, Sq, D_MODEL].
// ---------------------------------------------------------------------------
__global__ __launch_bounds__(32)
void attn_wmma_kernel(const float* __restrict__ Q, const float* __restrict__ Kp,
                      const float* __restrict__ Vp, float* __restrict__ O,
                      const float* __restrict__ fmask,
                      const int* __restrict__ toks, int Sq, int Sk, int mode) {
  __shared__ _Float16 pLds[16 * 32];
  const int qt = blockIdx.x, h = blockIdx.y, bi = blockIdx.z;
  const int lane = threadIdx.x;
  const int hg = lane >> 4, l15 = lane & 15;
  const int q0 = qt * 16;

  // Q tile in A layout, pre-scaled by 1/sqrt(DH)
  const float qscale = 0.17677669529663687f;
  v16h qa;
  const float* qrow = Q + ((size_t)(bi * Sq + q0 + l15)) * D_MODEL + h * D_HEAD;
#pragma unroll
  for (int i = 0; i < 16; ++i)
    qa[i] = (_Float16)(qrow[a_kidx(i, hg)] * qscale);

  float m_st[8], l_st[8];
#pragma unroll
  for (int j = 0; j < 8; ++j) { m_st[j] = -3.0e38f; l_st[j] = 0.f; }
  v8f o0 = {}, o1 = {};

  for (int kb = 0; kb < Sk; kb += 32) {
    // ---- scores for two 16-key subtiles ----
    v16h kt0, kt1;
#pragma unroll
    for (int i = 0; i < 16; ++i) {
      int dd = hg * 16 + i;   // B layout: K-dim = head dim here
      kt0[i] = (_Float16)Kp[((size_t)(bi * Sk + kb + l15)) * D_MODEL + h * D_HEAD + dd];
      kt1[i] = (_Float16)Kp[((size_t)(bi * Sk + kb + 16 + l15)) * D_MODEL + h * D_HEAD + dd];
    }
    v8f s0 = {}, s1 = {};
    s0 = wmma_f16(qa, kt0, s0);
    s1 = wmma_f16(qa, kt1, s1);

    // ---- additive mask, exactly as reference ----
    const int c0 = kb + l15, c1 = kb + 16 + l15;
    float add0 = 0.f, add1 = 0.f;
    if (mode == 0) {
      add0 = (1.f - fmask[bi * Sk + c0]) * -1e9f;
      add1 = (1.f - fmask[bi * Sk + c1]) * -1e9f;
    }
    float pd0 = 0.f, pd1 = 0.f;
    if (mode == 1) {
      pd0 = (toks[bi * Sk + c0] == 0) ? 1.f : 0.f;
      pd1 = (toks[bi * Sk + c1] == 0) ? 1.f : 0.f;
    }
#pragma unroll
    for (int j = 0; j < 8; ++j) {
      int row = q0 + j + hg * 8;
      if (mode == 0) {
        s0[j] += add0; s1[j] += add1;
      } else {
        float la0 = (c0 > row) ? 1.f : 0.f;
        float la1 = (c1 > row) ? 1.f : 0.f;
        s0[j] += (1.f - fmaxf(la0, pd0)) * -1e9f;
        s1[j] += (1.f - fmaxf(la1, pd1)) * -1e9f;
      }
    }

    // ---- online softmax update (row reductions stay inside half-wave) ----
    float pr0[8], pr1[8];
#pragma unroll
    for (int j = 0; j < 8; ++j) {
      float rm = fmaxf(s0[j], s1[j]);
#pragma unroll
      for (int off = 8; off; off >>= 1) rm = fmaxf(rm, __shfl_xor(rm, off, 32));
      float mn = fmaxf(m_st[j], rm);
      float alpha = __expf(m_st[j] - mn);
      pr0[j] = __expf(s0[j] - mn);
      pr1[j] = __expf(s1[j] - mn);
      float rs = pr0[j] + pr1[j];
#pragma unroll
      for (int off = 8; off; off >>= 1) rs += __shfl_xor(rs, off, 32);
      l_st[j] = l_st[j] * alpha + rs;
      m_st[j] = mn;
      o0[j] *= alpha;
      o1[j] *= alpha;
    }

    // ---- re-layout P (C layout -> A layout) through LDS ----
    __syncthreads();
#pragma unroll
    for (int j = 0; j < 8; ++j) {
      int row = j + hg * 8;
      pLds[row * 32 + l15]      = (_Float16)pr0[j];
      pLds[row * 32 + 16 + l15] = (_Float16)pr1[j];
    }
    __syncthreads();
    v16h pa;
#pragma unroll
    for (int i = 0; i < 16; ++i)
      pa[i] = pLds[l15 * 32 + a_kidx(i, hg)];

    // ---- P @ V: two 16-wide d tiles, K-dim = 32 keys ----
    v16h vb0, vb1;
#pragma unroll
    for (int i = 0; i < 16; ++i) {
      int key = kb + hg * 16 + i;
      const float* vr = Vp + ((size_t)(bi * Sk + key)) * D_MODEL + h * D_HEAD;
      vb0[i] = (_Float16)vr[l15];
      vb1[i] = (_Float16)vr[16 + l15];
    }
    o0 = wmma_f16(pa, vb0, o0);
    o1 = wmma_f16(pa, vb1, o1);
  }

  // ---- normalize + write head-merged output ----
#pragma unroll
  for (int j = 0; j < 8; ++j) {
    float inv = 1.f / l_st[j];
    int row = q0 + j + hg * 8;
    float* orow = O + ((size_t)(bi * Sq + row)) * D_MODEL + h * D_HEAD;
    orow[l15]      = o0[j] * inv;
    orow[16 + l15] = o1[j] * inv;
  }
}

// ---------------------------------------------------------------------------
// x = LayerNorm(x + a) * g + b ; one wave per 256-wide row, 8 rows per block.
// ---------------------------------------------------------------------------
__global__ __launch_bounds__(256)
void add_ln_kernel(float* __restrict__ x, const float* __restrict__ a,
                   const float* __restrict__ g, const float* __restrict__ bta,
                   int M) {
  const int wave = threadIdx.x >> 5, lane = threadIdx.x & 31;
  const int row = blockIdx.x * 8 + wave;
  if (row >= M) return;
  float* xr = x + (size_t)row * D_MODEL;
  const float* ar = a + (size_t)row * D_MODEL;
  float v[8];
  float sum = 0.f;
#pragma unroll
  for (int i = 0; i < 8; ++i) {
    v[i] = xr[lane + i * 32] + ar[lane + i * 32];
    sum += v[i];
  }
#pragma unroll
  for (int off = 16; off; off >>= 1) sum += __shfl_xor(sum, off, 32);
  float mu = sum * (1.f / 256.f);
  float vs = 0.f;
#pragma unroll
  for (int i = 0; i < 8; ++i) { float d = v[i] - mu; vs += d * d; }
#pragma unroll
  for (int off = 16; off; off >>= 1) vs += __shfl_xor(vs, off, 32);
  float inv = rsqrtf(vs * (1.f / 256.f) + 1e-6f);
#pragma unroll
  for (int i = 0; i < 8; ++i) {
    int d = lane + i * 32;
    xr[d] = g[d] * (v[i] - mu) * inv + bta[d];
  }
}

// ---------------------------------------------------------------------------
// Host-side orchestration.
// ---------------------------------------------------------------------------
static inline void launch_gemm(const float* A, const float* W, const float* bias,
                               float* C, int M, int N, int K, int relu,
                               hipStream_t s) {
  dim3 grid((N + BN - 1) / BN, M / 16);
  gemm_wmma_kernel<<<grid, 256, 0, s>>>(A, W, bias, C, M, N, K, relu);
}

extern "C" void kernel_launch(void* const* d_in, const int* in_sizes, int n_in,
                              void* d_out, int out_size, void* d_ws,
                              size_t ws_size, hipStream_t stream) {
  (void)in_sizes; (void)n_in; (void)out_size; (void)ws_size;
  const int B = 4, SE = 128, SD = 2048, L = 2;
  const int Dm = 256, F = 512, VD = 10002;

  // ---- inputs in setup_inputs() dict order ----
  const int*   enc_inputs = (const int*)  d_in[0];
  const float* enc_mask   = (const float*)d_in[1];   // [B,1,1,SE], 1 = attend
  const int*   dec_inputs = (const int*)  d_in[2];
  const float* enc_emb    = (const float*)d_in[3];
  const float* dec_emb    = (const float*)d_in[4];
  const float* out_w      = (const float*)d_in[5];
  const float* out_b      = (const float*)d_in[6];
  const float* W_qkvo[3][4];   // [enc,dec1,dec2][wq,wk,wv,wo]
  const float* B_qkvo[3][4];
  for (int p = 0; p < 3; ++p) {
    for (int n = 0; n < 4; ++n) W_qkvo[p][n] = (const float*)d_in[7 + p * 8 + n];
    for (int n = 0; n < 4; ++n) B_qkvo[p][n] = (const float*)d_in[7 + p * 8 + 4 + n];
  }
  const float* enc_w1 = (const float*)d_in[31];
  const float* enc_b1 = (const float*)d_in[32];
  const float* enc_w2 = (const float*)d_in[33];
  const float* enc_b2 = (const float*)d_in[34];
  const float* dec_w1 = (const float*)d_in[35];
  const float* dec_b1 = (const float*)d_in[36];
  const float* dec_w2 = (const float*)d_in[37];
  const float* dec_b2 = (const float*)d_in[38];
  const float* ln[5][2];  // enc_ln1, enc_ln2, dec_ln1, dec_ln2, dec_ln3 (g,b)
  for (int i = 0; i < 5; ++i) {
    ln[i][0] = (const float*)d_in[39 + 2 * i];
    ln[i][1] = (const float*)d_in[39 + 2 * i + 1];
  }

  // ---- workspace carve-up (~68 MB of f32) ----
  float* w = (float*)d_ws;
  float* xenc = w; w += (size_t)B * SE * Dm;   // encoder activations / code
  float* ybuf = w; w += (size_t)B * SD * Dm;   // decoder activations
  float* bq   = w; w += (size_t)B * SD * Dm;
  float* bk   = w; w += (size_t)B * SD * Dm;
  float* bv   = w; w += (size_t)B * SD * Dm;
  float* ba   = w; w += (size_t)B * SD * Dm;
  float* bt   = w; w += (size_t)B * SD * Dm;
  float* bf   = w; w += (size_t)B * SD * F;

  // ================= encoder =================
  {
    int tot = B * SE * Dm;
    embed_pos_kernel<<<tot / 256, 256, 0, stream>>>(enc_inputs, enc_emb, xenc, SE, tot);
  }
  for (int l = 0; l < L; ++l) {
    const size_t wofs = (size_t)l * Dm * Dm, bofs = (size_t)l * Dm;
    launch_gemm(xenc, W_qkvo[0][0] + wofs, B_qkvo[0][0] + bofs, bq, B * SE, Dm, Dm, 0, stream);
    launch_gemm(xenc, W_qkvo[0][1] + wofs, B_qkvo[0][1] + bofs, bk, B * SE, Dm, Dm, 0, stream);
    launch_gemm(xenc, W_qkvo[0][2] + wofs, B_qkvo[0][2] + bofs, bv, B * SE, Dm, Dm, 0, stream);
    attn_wmma_kernel<<<dim3(SE / 16, 8, B), 32, 0, stream>>>(bq, bk, bv, ba,
        enc_mask, nullptr, SE, SE, 0);
    launch_gemm(ba, W_qkvo[0][3] + wofs, B_qkvo[0][3] + bofs, bt, B * SE, Dm, Dm, 0, stream);
    add_ln_kernel<<<(B * SE) / 8, 256, 0, stream>>>(xenc, bt, ln[0][0] + bofs, ln[0][1] + bofs, B * SE);
    launch_gemm(xenc, enc_w1 + (size_t)l * Dm * F, enc_b1 + (size_t)l * F, bf, B * SE, F, Dm, 1, stream);
    launch_gemm(bf, enc_w2 + (size_t)l * F * Dm, enc_b2 + bofs, bt, B * SE, Dm, F, 0, stream);
    add_ln_kernel<<<(B * SE) / 8, 256, 0, stream>>>(xenc, bt, ln[1][0] + bofs, ln[1][1] + bofs, B * SE);
  }

  // ================= decoder =================
  {
    int tot = B * SD * Dm;
    embed_pos_kernel<<<tot / 256, 256, 0, stream>>>(dec_inputs, dec_emb, ybuf, SD, tot);
  }
  for (int l = 0; l < L; ++l) {
    const size_t wofs = (size_t)l * Dm * Dm, bofs = (size_t)l * Dm;
    // -- masked self-attention --
    launch_gemm(ybuf, W_qkvo[1][0] + wofs, B_qkvo[1][0] + bofs, bq, B * SD, Dm, Dm, 0, stream);
    launch_gemm(ybuf, W_qkvo[1][1] + wofs, B_qkvo[1][1] + bofs, bk, B * SD, Dm, Dm, 0, stream);
    launch_gemm(ybuf, W_qkvo[1][2] + wofs, B_qkvo[1][2] + bofs, bv, B * SD, Dm, Dm, 0, stream);
    attn_wmma_kernel<<<dim3(SD / 16, 8, B), 32, 0, stream>>>(bq, bk, bv, ba,
        nullptr, dec_inputs, SD, SD, 1);
    launch_gemm(ba, W_qkvo[1][3] + wofs, B_qkvo[1][3] + bofs, bt, B * SD, Dm, Dm, 0, stream);
    add_ln_kernel<<<(B * SD) / 8, 256, 0, stream>>>(ybuf, bt, ln[2][0] + bofs, ln[2][1] + bofs, B * SD);
    // -- cross-attention (K/V from encoder code) --
    launch_gemm(ybuf, W_qkvo[2][0] + wofs, B_qkvo[2][0] + bofs, bq, B * SD, Dm, Dm, 0, stream);
    launch_gemm(xenc, W_qkvo[2][1] + wofs, B_qkvo[2][1] + bofs, bk, B * SE, Dm, Dm, 0, stream);
    launch_gemm(xenc, W_qkvo[2][2] + wofs, B_qkvo[2][2] + bofs, bv, B * SE, Dm, Dm, 0, stream);
    attn_wmma_kernel<<<dim3(SD / 16, 8, B), 32, 0, stream>>>(bq, bk, bv, ba,
        enc_mask, nullptr, SD, SE, 0);
    launch_gemm(ba, W_qkvo[2][3] + wofs, B_qkvo[2][3] + bofs, bt, B * SD, Dm, Dm, 0, stream);
    add_ln_kernel<<<(B * SD) / 8, 256, 0, stream>>>(ybuf, bt, ln[3][0] + bofs, ln[3][1] + bofs, B * SD);
    // -- FFN --
    launch_gemm(ybuf, dec_w1 + (size_t)l * Dm * F, dec_b1 + (size_t)l * F, bf, B * SD, F, Dm, 1, stream);
    launch_gemm(bf, dec_w2 + (size_t)l * F * Dm, dec_b2 + bofs, bt, B * SD, Dm, F, 0, stream);
    add_ln_kernel<<<(B * SD) / 8, 256, 0, stream>>>(ybuf, bt, ln[4][0] + bofs, ln[4][1] + bofs, B * SD);
  }

  // ================= final vocab projection =================
  launch_gemm(ybuf, out_w, out_b, (float*)d_out, B * SD, VD, Dm, 0, stream);
}
